// Tacotron_19688130085875
// MI455X (gfx1250) — compile-verified, matrix operand order checked by reading, not run on hardware
//
#include <hip/hip_runtime.h>
#include <hip/hip_bf16.h>

// ---------------- constants (match reference) ----------------
#define NUM_MELS 80
#define EMB      512
#define ATT_RNN  1024
#define DEC_RNN  1024
#define PRENET   256
#define ATT_DIM  128
#define LOC_F    32
#define LOC_K    31
#define BB       32
#define T_ENC    256
#define T_DEC    256
#define NEG_INF  -1000000000.0f

// Concatenated-K sizes for the fused LSTM GEMMs
#define KA (PRENET + EMB + ATT_RNN)      // 1792  : [pre | ctx | h_a]
#define KD (ATT_RNN + EMB + DEC_RNN)     // 2560  : [h_a | ctx | h_d]

// LDS row padding (bf16 elements): keeps rows 32B-aligned and staggers banks
#define APAD 16

typedef __attribute__((ext_vector_type(16))) __bf16 v16bf;
typedef __attribute__((ext_vector_type(8)))  float  v8f;

__device__ __forceinline__ float sigf(float x) { return 1.0f / (1.0f + __expf(-x)); }

// =====================================================================
// WMMA bf16 GEMM:  Z[M x N] = X[M x K] (row stride lda) @ W[N x K]^T
// blockDim.x = 256 (8 waves). Each wave -> one 32x32 C tile.
// grid.x covers N in chunks of 256, grid.y covers M in chunks of 32.
// The 32xK activation block is staged once per WG into LDS using
// GLOBAL_LOAD_ASYNC_TO_LDS_B128 (ASYNCcnt), then read via ds_load for
// the WMMA A fragments; B (weights) streams from global/L2.
// Dynamic LDS: 32*(K+APAD)*2 bytes.
// =====================================================================
__global__ void __launch_bounds__(256)
gemm_bf16_wmma(const __bf16* __restrict__ X, int lda,
               const __bf16* __restrict__ W,
               float* __restrict__ Z, int N, int K)
{
    extern __shared__ __bf16 sX[];              // 32 x (K+APAD)
    const int tid  = threadIdx.x;
    const int m0   = blockIdx.y * 32;
    const int krow = K + APAD;

    // ---- stage X block (rows m0..m0+31, K cols) into LDS, async ----
    {
        const int cpr = K >> 3;                 // 16B chunks per row
        for (int i = tid; i < 4 * K; i += 256) {
            int row = i / cpr, c = i - row * cpr;
            const __bf16* g = X + (size_t)(m0 + row) * lda + (size_t)c * 8;
            unsigned ldsoff = (unsigned)(size_t)(sX + (size_t)row * krow + (size_t)c * 8);
            asm volatile("global_load_async_to_lds_b128 %0, %1, off"
                         :: "v"(ldsoff), "v"(g) : "memory");
        }
        asm volatile("s_wait_asynccnt 0" ::: "memory");
    }
    __syncthreads();

    const int wave = tid >> 5;
    const int lane = tid & 31;
    const int n0 = (blockIdx.x * 8 + wave) * 32;
    if (n0 >= N) return;                        // wave-uniform (after barrier)

    v8f acc00 = {}, acc01 = {}, acc10 = {}, acc11 = {};

    // A layout: lane (L&15) = M row, (L>>4) selects K half (16 bf16 contiguous)
    const int r  = lane & 15;
    const int kh = (lane >> 4) * 16;
    const __bf16* la0 = sX + (size_t)r        * krow + kh;   // rows 0..15
    const __bf16* la1 = sX + (size_t)(16 + r) * krow + kh;   // rows 16..31
    // B layout: lane = N column, holds 16 contiguous K values (W row-major NxK)
    const __bf16* wb0 = W + (size_t)(n0 + r)      * K + kh;
    const __bf16* wb1 = W + (size_t)(n0 + 16 + r) * K + kh;

    for (int k = 0; k < K; k += 32) {
        v16bf a0 = *(const v16bf*)(la0 + k);    // ds_load_b128 x2
        v16bf a1 = *(const v16bf*)(la1 + k);
        v16bf b0 = *(const v16bf*)(wb0 + k);
        v16bf b1 = *(const v16bf*)(wb1 + k);
        __builtin_prefetch(wb0 + k + 256, 0, 1);   // global_prefetch on weight stream
        acc00 = __builtin_amdgcn_wmma_f32_16x16x32_bf16(false, a0, false, b0, (short)0, acc00, false, false);
        acc01 = __builtin_amdgcn_wmma_f32_16x16x32_bf16(false, a0, false, b1, (short)0, acc01, false, false);
        acc10 = __builtin_amdgcn_wmma_f32_16x16x32_bf16(false, a1, false, b0, (short)0, acc10, false, false);
        acc11 = __builtin_amdgcn_wmma_f32_16x16x32_bf16(false, a1, false, b1, (short)0, acc11, false, false);
    }

    // C/D layout: VGPR v, lanes 0-15: (M=v, N=lane); lanes 16-31: (M=8+v, N=lane-16)
    const int crow = (lane >> 4) * 8;
    const int ccol = lane & 15;
    #pragma unroll
    for (int v = 0; v < 8; ++v) {
        const size_t mr0 = (size_t)(m0 + crow + v);
        const size_t mr1 = (size_t)(m0 + 16 + crow + v);
        Z[mr0 * N + n0 + ccol]      = acc00[v];
        Z[mr0 * N + n0 + 16 + ccol] = acc01[v];
        Z[mr1 * N + n0 + ccol]      = acc10[v];
        Z[mr1 * N + n0 + 16 + ccol] = acc11[v];
    }
}

// =====================================================================
// Precompute kernels
// =====================================================================
__global__ void cvt_f32_bf16(const float* __restrict__ src, __bf16* __restrict__ dst, int n)
{
    int i = blockIdx.x * blockDim.x + threadIdx.x;
    if (i < n) dst[i] = (__bf16)src[i];
}

__global__ void relu_cvt_bf16(const float* __restrict__ src, __bf16* __restrict__ dst, int n)
{
    int i = blockIdx.x * blockDim.x + threadIdx.x;
    if (i < n) { float v = src[i]; dst[i] = (__bf16)(v > 0.f ? v : 0.f); }
}

// Build [Wih | Whh] concatenated bf16 weight, rows x (Kih+Khh)
__global__ void build_wcat(const float* __restrict__ Wih, int Kih,
                           const float* __restrict__ Whh, int Khh,
                           __bf16* __restrict__ dst, int n)
{
    int i = blockIdx.x * blockDim.x + threadIdx.x;
    if (i >= n) return;
    const int K = Kih + Khh;
    int row = i / K, k = i % K;
    float v = (k < Kih) ? Wih[(size_t)row * Kih + k] : Whh[(size_t)row * Khh + (k - Kih)];
    dst[i] = (__bf16)v;
}

// Prenet layer 1: h1[(t*BB+b), j] = relu( dec_in[t,b,:] @ W1[j,:] ), dec_in[0]=0
__global__ void prenet1(const float* __restrict__ inputs /*B,T,80*/,
                        const float* __restrict__ W1 /*256x80*/,
                        __bf16* __restrict__ h1 /*8192 x 256*/)
{
    int rrow = blockIdx.x;            // 0..8191 = t*32+b
    int j    = threadIdx.x;           // 0..255
    int t = rrow >> 5, b = rrow & 31;
    float acc = 0.f;
    if (t > 0) {
        const float* x = inputs + ((size_t)b * T_DEC + (t - 1)) * NUM_MELS;
        const float* w = W1 + (size_t)j * NUM_MELS;
        #pragma unroll 8
        for (int k = 0; k < NUM_MELS; ++k) acc += x[k] * w[k];
    }
    h1[(size_t)rrow * PRENET + j] = (__bf16)(acc > 0.f ? acc : 0.f);
}

// Per-step: copy prenet row t into X_a[:, 0:256]
__global__ void copy_pre(const __bf16* __restrict__ pre_bf, int t, __bf16* __restrict__ Xa)
{
    int b = blockIdx.x, c = threadIdx.x;  // grid 32, block 256
    Xa[(size_t)b * KA + c] = pre_bf[((size_t)t * BB + b) * PRENET + c];
}

// =====================================================================
// LSTM pointwise: gates from Z (32 x 4H) + bias -> h,c ; also scatter bf16 h
// =====================================================================
__global__ void lstm_pointwise(const float* __restrict__ Z, const float* __restrict__ bias,
                               float* __restrict__ c_state, float* __restrict__ h_state,
                               __bf16* __restrict__ hb0, int ld0,
                               __bf16* __restrict__ hb1, int ld1, int H)
{
    int i = blockIdx.x * blockDim.x + threadIdx.x;
    if (i >= BB * H) return;
    int b = i / H, j = i % H;
    const float* zr = Z + (size_t)b * 4 * H;
    float zi = zr[j]         + bias[j];
    float zf = zr[H + j]     + bias[H + j];
    float zg = zr[2 * H + j] + bias[2 * H + j];
    float zo = zr[3 * H + j] + bias[3 * H + j];
    float c  = sigf(zf) * c_state[i] + sigf(zi) * tanhf(zg);
    float h  = sigf(zo) * tanhf(c);
    c_state[i] = c;
    h_state[i] = h;
    __bf16 hb = (__bf16)h;
    hb0[(size_t)b * ld0 + j] = hb;
    if (hb1) hb1[(size_t)b * ld1 + j] = hb;
}

// =====================================================================
// Attention energies: location conv + tanh(pq + loc + pm) . v, masked
// grid 32 (b), block 256 (t)
// =====================================================================
__global__ void attention_energies(const float* __restrict__ att_w,
                                   const float* __restrict__ att_w_cum,
                                   const float* __restrict__ Wconv /*32x2x31*/,
                                   const float* __restrict__ Wld   /*128x32*/,
                                   const float* __restrict__ vatt  /*128*/,
                                   const float* __restrict__ pq    /*32x128*/,
                                   const float* __restrict__ pm    /*8192x128*/,
                                   const int*   __restrict__ mem_len,
                                   float* __restrict__ energies /*32x256*/)
{
    int b = blockIdx.x, t = threadIdx.x;
    float loc[LOC_F];
    #pragma unroll
    for (int f = 0; f < LOC_F; ++f) loc[f] = 0.f;

    const float* srcs[2] = { att_w + (size_t)b * T_ENC, att_w_cum + (size_t)b * T_ENC };
    for (int c = 0; c < 2; ++c) {
        const float* s = srcs[c];
        for (int k = 0; k < LOC_K; ++k) {
            int tt = t + k - (LOC_K - 1) / 2;
            if (tt < 0 || tt >= T_ENC) continue;
            float x = s[tt];
            #pragma unroll
            for (int f = 0; f < LOC_F; ++f)
                loc[f] += x * Wconv[(f * 2 + c) * LOC_K + k];
        }
    }

    const float* pqr = pq + (size_t)b * ATT_DIM;
    const float* pmr = pm + ((size_t)b * T_ENC + t) * ATT_DIM;
    float e = 0.f;
    for (int j = 0; j < ATT_DIM; ++j) {
        float s = pqr[j] + pmr[j];
        const float* wl = Wld + (size_t)j * LOC_F;
        #pragma unroll
        for (int f = 0; f < LOC_F; ++f) s += wl[f] * loc[f];
        e += tanhf(s) * vatt[j];
    }
    if (t >= mem_len[b]) e = NEG_INF;
    energies[(size_t)b * T_ENC + t] = e;
}

// Softmax over T_enc, update cumulative, write alignment row of d_out
__global__ void softmax_cum(const float* __restrict__ energies,
                            float* __restrict__ att_w, float* __restrict__ att_w_cum,
                            float* __restrict__ out_align, int step)
{
    __shared__ float red[256];
    int b = blockIdx.x, t = threadIdx.x;
    float e = energies[(size_t)b * T_ENC + t];
    red[t] = e; __syncthreads();
    for (int s = 128; s > 0; s >>= 1) {
        if (t < s) red[t] = fmaxf(red[t], red[t + s]);
        __syncthreads();
    }
    float mx = red[0]; __syncthreads();
    float ex = __expf(e - mx);
    red[t] = ex; __syncthreads();
    for (int s = 128; s > 0; s >>= 1) {
        if (t < s) red[t] += red[t + s];
        __syncthreads();
    }
    float w = ex / red[0];
    size_t i = (size_t)b * T_ENC + t;
    att_w[i] = w;
    att_w_cum[i] += w;
    out_align[((size_t)b * T_DEC + step) * T_ENC + t] = w;
}

// Context: ctx[b,d] = sum_t w[t] * memory[b,t,d]; scatter bf16 ctx into X_a/X_d
__global__ void context(const float* __restrict__ att_w, const float* __restrict__ memory,
                        float* __restrict__ ctx, __bf16* __restrict__ Xa, __bf16* __restrict__ Xd)
{
    __shared__ float w[T_ENC];
    int b = blockIdx.x, tid = threadIdx.x;  // block 256
    w[tid] = att_w[(size_t)b * T_ENC + tid];
    __syncthreads();
    const float* mb = memory + (size_t)b * T_ENC * EMB;
    for (int d = tid; d < EMB; d += 256) {
        float acc = 0.f;
        for (int t = 0; t < T_ENC; ++t) acc += w[t] * mb[(size_t)t * EMB + d];
        ctx[(size_t)b * EMB + d] = acc;
        __bf16 cb = (__bf16)acc;
        Xa[(size_t)b * KA + PRENET + d]  = cb;
        Xd[(size_t)b * KD + ATT_RNN + d] = cb;
    }
}

// Projection: mel (80) + gate (1) from hc = [h_d | ctx]
__global__ void projection(const float* __restrict__ h_d, const float* __restrict__ ctx,
                           const float* __restrict__ Wp /*80x1536*/, const float* __restrict__ bp,
                           const float* __restrict__ Wg /*1x1536*/, const float* __restrict__ bg,
                           float* __restrict__ out_mel, float* __restrict__ out_gate, int step)
{
    int b = blockIdx.x, j = threadIdx.x;  // block 128
    if (j > NUM_MELS) return;
    const float* hd = h_d + (size_t)b * DEC_RNN;
    const float* cx = ctx + (size_t)b * EMB;
    const float* w  = (j < NUM_MELS) ? (Wp + (size_t)j * (DEC_RNN + EMB)) : Wg;
    float acc = (j < NUM_MELS) ? bp[j] : bg[0];
    for (int k = 0; k < DEC_RNN; ++k)      acc += w[k] * hd[k];
    for (int k = 0; k < EMB; ++k)          acc += w[DEC_RNN + k] * cx[k];
    if (j < NUM_MELS)
        out_mel[((size_t)b * T_DEC + step) * NUM_MELS + j] = acc;
    else
        out_gate[(size_t)b * T_DEC + step] = acc;
}

// =====================================================================
// Host side
// =====================================================================
static inline size_t gemm_lds_bytes(int K) { return (size_t)32 * (K + APAD) * 2; }

extern "C" void kernel_launch(void* const* d_in, const int* in_sizes, int n_in,
                              void* d_out, int out_size, void* d_ws, size_t ws_size,
                              hipStream_t stream)
{
    const float* enc      = (const float*)d_in[0];   // 32,256,512
    const float* inputs   = (const float*)d_in[1];   // 32,256,80
    const int*   mem_len  = (const int*)  d_in[2];
    const float* W_pre1   = (const float*)d_in[3];
    const float* W_pre2   = (const float*)d_in[4];
    const float* Wih_a    = (const float*)d_in[5];
    const float* Whh_a    = (const float*)d_in[6];
    const float* b_a      = (const float*)d_in[7];
    const float* W_q      = (const float*)d_in[8];
    const float* W_mem    = (const float*)d_in[9];
    const float* W_lconv  = (const float*)d_in[10];
    const float* W_ldense = (const float*)d_in[11];
    const float* v_att    = (const float*)d_in[12];
    const float* Wih_d    = (const float*)d_in[13];
    const float* Whh_d    = (const float*)d_in[14];
    const float* b_d      = (const float*)d_in[15];
    const float* W_proj   = (const float*)d_in[16];
    const float* b_proj   = (const float*)d_in[17];
    const float* W_gate   = (const float*)d_in[18];
    const float* b_gate   = (const float*)d_in[19];

    float* out_mel   = (float*)d_out;
    float* out_gate  = out_mel + (size_t)BB * T_DEC * NUM_MELS;
    float* out_align = out_gate + (size_t)BB * T_DEC;

    // -------- workspace layout --------
    char* ws = (char*)d_ws;
    size_t off = 0;
    auto alloc = [&](size_t bytes) -> char* {
        char* p = ws + off;
        off = (off + bytes + 255) & ~(size_t)255;
        return p;
    };
    __bf16* Wa_cat  = (__bf16*)alloc((size_t)4 * ATT_RNN * KA * 2);
    __bf16* Wd_cat  = (__bf16*)alloc((size_t)4 * DEC_RNN * KD * 2);
    __bf16* Wq_bf   = (__bf16*)alloc((size_t)ATT_DIM * ATT_RNN * 2);
    __bf16* Wp2_bf  = (__bf16*)alloc((size_t)PRENET * PRENET * 2);
    __bf16* Wm_bf   = (__bf16*)alloc((size_t)ATT_DIM * EMB * 2);
    __bf16* mem_bf  = (__bf16*)alloc((size_t)BB * T_ENC * EMB * 2);
    __bf16* h1_bf   = (__bf16*)alloc((size_t)T_DEC * BB * PRENET * 2);
    float*  pre_f   = (float*) alloc((size_t)T_DEC * BB * PRENET * 4);
    __bf16* pre_bf  = (__bf16*)alloc((size_t)T_DEC * BB * PRENET * 2);
    float*  pm      = (float*) alloc((size_t)BB * T_ENC * ATT_DIM * 4);
    float*  Z_a     = (float*) alloc((size_t)BB * 4 * ATT_RNN * 4);
    float*  Z_d     = (float*) alloc((size_t)BB * 4 * DEC_RNN * 4);
    float*  pq      = (float*) alloc((size_t)BB * ATT_DIM * 4);
    float*  energ   = (float*) alloc((size_t)BB * T_ENC * 4);
    // zero-initialized state block (contiguous)
    char*   state0  = ws + off;
    float*  h_a     = (float*) alloc((size_t)BB * ATT_RNN * 4);
    float*  c_a     = (float*) alloc((size_t)BB * ATT_RNN * 4);
    float*  h_d     = (float*) alloc((size_t)BB * DEC_RNN * 4);
    float*  c_d     = (float*) alloc((size_t)BB * DEC_RNN * 4);
    float*  ctx     = (float*) alloc((size_t)BB * EMB * 4);
    float*  att_w   = (float*) alloc((size_t)BB * T_ENC * 4);
    float*  att_wc  = (float*) alloc((size_t)BB * T_ENC * 4);
    __bf16* X_a     = (__bf16*)alloc((size_t)BB * KA * 2);
    __bf16* X_d     = (__bf16*)alloc((size_t)BB * KD * 2);
    size_t stateBytes = (size_t)((ws + off) - state0);

    // -------- init state (graph-capturable) --------
    hipMemsetAsync(state0, 0, stateBytes, stream);

    // -------- one-time weight prep --------
    {
        int n = 4 * ATT_RNN * KA;
        build_wcat<<<(n + 255) / 256, 256, 0, stream>>>(Wih_a, PRENET + EMB, Whh_a, ATT_RNN, Wa_cat, n);
        n = 4 * DEC_RNN * KD;
        build_wcat<<<(n + 255) / 256, 256, 0, stream>>>(Wih_d, ATT_RNN + EMB, Whh_d, DEC_RNN, Wd_cat, n);
        n = ATT_DIM * ATT_RNN;
        cvt_f32_bf16<<<(n + 255) / 256, 256, 0, stream>>>(W_q, Wq_bf, n);
        n = PRENET * PRENET;
        cvt_f32_bf16<<<(n + 255) / 256, 256, 0, stream>>>(W_pre2, Wp2_bf, n);
        n = ATT_DIM * EMB;
        cvt_f32_bf16<<<(n + 255) / 256, 256, 0, stream>>>(W_mem, Wm_bf, n);
        n = BB * T_ENC * EMB;
        cvt_f32_bf16<<<(n + 255) / 256, 256, 0, stream>>>(enc, mem_bf, n);
    }

    // -------- prenet: layer1 (plain), layer2 (WMMA, M=8192), relu+cvt --------
    prenet1<<<T_DEC * BB, PRENET, 0, stream>>>(inputs, W_pre1, h1_bf);
    {
        dim3 g((PRENET + 255) / 256, (T_DEC * BB) / 32);
        gemm_bf16_wmma<<<g, 256, gemm_lds_bytes(PRENET), stream>>>(h1_bf, PRENET, Wp2_bf, pre_f, PRENET, PRENET);
        int n = T_DEC * BB * PRENET;
        relu_cvt_bf16<<<(n + 255) / 256, 256, 0, stream>>>(pre_f, pre_bf, n);
    }
    // -------- processed_memory = memory @ W_mem^T (WMMA, M=8192, N=128, K=512) --------
    {
        dim3 g(1, (BB * T_ENC) / 32);
        gemm_bf16_wmma<<<g, 256, gemm_lds_bytes(EMB), stream>>>(mem_bf, EMB, Wm_bf, pm, ATT_DIM, EMB);
    }

    // -------- the sequential scan --------
    for (int t = 0; t < T_DEC; ++t) {
        // X_a[:,0:256] = pre[t]
        copy_pre<<<BB, PRENET, 0, stream>>>(pre_bf, t, X_a);

        // attention LSTM GEMM: Z_a = X_a @ [Wih_a|Whh_a]^T   (N=4096, K=1792)
        {
            dim3 g((4 * ATT_RNN) / 256, 1);
            gemm_bf16_wmma<<<g, 256, gemm_lds_bytes(KA), stream>>>(X_a, KA, Wa_cat, Z_a, 4 * ATT_RNN, KA);
        }
        // pointwise -> h_a, c_a; bf16 h_a -> X_a[:,768:] and X_d[:,0:1024]
        lstm_pointwise<<<(BB * ATT_RNN + 255) / 256, 256, 0, stream>>>(
            Z_a, b_a, c_a, h_a, X_a + (PRENET + EMB), KA, X_d, KD, ATT_RNN);

        // pq = h_a @ W_q^T  (A = X_d cols 0..1023 as bf16, lda=KD; N=128, K=1024)
        {
            dim3 g(1, 1);
            gemm_bf16_wmma<<<g, 256, gemm_lds_bytes(ATT_RNN), stream>>>(X_d, KD, Wq_bf, pq, ATT_DIM, ATT_RNN);
        }

        // energies / softmax / context
        attention_energies<<<BB, T_ENC, 0, stream>>>(att_w, att_wc, W_lconv, W_ldense,
                                                     v_att, pq, pm, mem_len, energ);
        softmax_cum<<<BB, T_ENC, 0, stream>>>(energ, att_w, att_wc, out_align, t);
        context<<<BB, 256, 0, stream>>>(att_w, enc, ctx, X_a, X_d);

        // decoder LSTM GEMM: Z_d = X_d @ [Wih_d|Whh_d]^T   (N=4096, K=2560)
        {
            dim3 g((4 * DEC_RNN) / 256, 1);
            gemm_bf16_wmma<<<g, 256, gemm_lds_bytes(KD), stream>>>(X_d, KD, Wd_cat, Z_d, 4 * DEC_RNN, KD);
        }
        // pointwise -> h_d, c_d; bf16 h_d -> X_d[:,1536:]
        lstm_pointwise<<<(BB * DEC_RNN + 255) / 256, 256, 0, stream>>>(
            Z_d, b_d, c_d, h_d, X_d + (ATT_RNN + EMB), KD, (__bf16*)nullptr, 0, DEC_RNN);

        // projection: mel + gate
        projection<<<BB, 128, 0, stream>>>(h_d, ctx, W_proj, b_proj, W_gate, b_gate,
                                           out_mel, out_gate, t);
    }
}
